// TopKDecoder_58703613002499
// MI455X (gfx1250) — matrix operand order repaired and checked by reading, not compiled, not run on hardware
//
#include <hip/hip_runtime.h>
#include <hip/hip_bf16.h>
#include <math.h>

// ---------------- problem constants ----------------
#define BSZ   16
#define BEAMK 8
#define NROWS 128          // BSZ*BEAMK
#define HDIM  1024
#define EDIM  1024
#define VDIM  32000
#define LSEQ  64
#define TMAX  64
#define NEGF  (-1.0e30f)
#define SOS_ID 1
#define EOS_ID 2

typedef __bf16  v16bf __attribute__((ext_vector_type(16)));
typedef float   v8f   __attribute__((ext_vector_type(8)));

__device__ __forceinline__ void wait_async0() {
#if __has_builtin(__builtin_amdgcn_s_wait_asynccnt)
    __builtin_amdgcn_s_wait_asynccnt(0);
#else
    asm volatile("s_wait_asynccnt 0x0" ::: "memory");
#endif
}

// =====================================================================
// fp32-in / fp32-out GEMM, bf16 WMMA compute.
// C[M x Nd] = act( A[M x Kd] @ B[Kd x Nd] + bias )
// block = 128 threads (4 waves); block tile = 16(M) x 256(N);
// each wave owns a 16x64 strip -> 4 x v_wmma_f32_16x16x32_bf16 per K-chunk.
// B chunk (32x256 fp32) streamed into LDS with global_load_async_to_lds_b128
// (ASYNCcnt), A chunk staged pre-permuted as bf16 fragments (one contiguous
// 32B ds load per lane).  grid = (Nd/256, M/16).
// =====================================================================
#define BN_PAD 264   // 256 + 8 floats pad (row = 1056B, 16B aligned)

__global__ __launch_bounds__(128)
void gemm_wmma_bf16(const float* __restrict__ A, const float* __restrict__ Bm,
                    const float* __restrict__ bias, float* __restrict__ C,
                    int Kd, int Nd, int act)
{
    __shared__ __align__(16) float  Bs[32][BN_PAD];   // ~33 KB
    __shared__ __align__(32) __bf16 As_frag[512];     // 1 KB, fragment order

    const int tid  = threadIdx.x;
    const int lane = tid & 31;
    const int wave = tid >> 5;
    const int nb0  = blockIdx.x * 256;        // block N base
    const int m0   = blockIdx.y * 16;         // block M base

    // B fragment mapping (16-bit B 32x16, ISA 7.12.2): lanes 0-15 hold
    // N=lane, K in {0..7,16..23}; lanes 16-31 hold N=lane-16, K in {8..15,24..31}.
    const int bn     = lane & 15;
    const int bkbase = (lane >> 4) * 8;

    v8f acc0 = {}, acc1 = {}, acc2 = {}, acc3 = {};

    for (int k0 = 0; k0 < Kd; k0 += 32) {
        __syncthreads();   // previous chunk fully consumed before overwrite

        // ---- stage A tile (16x32) as bf16 in WMMA-A fragment order ----
        // element i = frag_lane*16 + j ; lanes 0-15: M=lane, K in
        // {0..7,16..23}; lanes 16-31: M=lane-16, K in {8..15,24..31}.
        for (int i = tid; i < 512; i += 128) {
            int fl = i >> 4, j = i & 15;
            int m = fl & 15, kh = fl >> 4;
            int k = (j < 8) ? (kh * 8 + j) : (16 + kh * 8 + (j - 8));
            As_frag[i] = (__bf16)A[(size_t)(m0 + m) * Kd + k0 + k];
        }

        // ---- async-stream B chunk 32x256 fp32 into LDS (coalesced b128) ----
        for (int g = tid; g < 2048; g += 128) {
            int k  = g >> 6;
            int nq = (g & 63) << 2;
            unsigned int lds_off = (unsigned int)(size_t)&Bs[k][nq];
            const float* gp = Bm + (size_t)(k0 + k) * Nd + nb0 + nq;
            asm volatile("global_load_async_to_lds_b128 %0, %1, off"
                         :: "v"(lds_off), "v"(gp) : "memory");
        }
        wait_async0();
        __syncthreads();

        // ---- fragments + 4 WMMA ----
        v16bf afrag = *(const v16bf*)&As_frag[lane * 16];
        const int nl0 = wave * 64 + bn;
#pragma unroll
        for (int tn = 0; tn < 4; ++tn) {
            v16bf bfrag;
#pragma unroll
            for (int j = 0; j < 16; ++j)
                bfrag[j] = (__bf16)Bs[bkbase + j + (j & 8)][nl0 + tn * 16];
            switch (tn) {
            case 0: acc0 = __builtin_amdgcn_wmma_f32_16x16x32_bf16(
                        false, afrag, false, bfrag, (short)0, acc0, false, false); break;
            case 1: acc1 = __builtin_amdgcn_wmma_f32_16x16x32_bf16(
                        false, afrag, false, bfrag, (short)0, acc1, false, false); break;
            case 2: acc2 = __builtin_amdgcn_wmma_f32_16x16x32_bf16(
                        false, afrag, false, bfrag, (short)0, acc2, false, false); break;
            default: acc3 = __builtin_amdgcn_wmma_f32_16x16x32_bf16(
                        false, afrag, false, bfrag, (short)0, acc3, false, false); break;
            }
        }
    }

    // ---- epilogue: C/D layout (VGPR r: lanes 0-15 M=r, lanes 16-31 M=8+r) ----
    const int rbase = m0 + ((lane >> 4) << 3);
#pragma unroll
    for (int tn = 0; tn < 4; ++tn) {
        const v8f acc = (tn == 0) ? acc0 : (tn == 1) ? acc1 : (tn == 2) ? acc2 : acc3;
        const int col = nb0 + wave * 64 + tn * 16 + (lane & 15);
#pragma unroll
        for (int r = 0; r < 8; ++r) {
            float v = acc[r];
            if (bias) v += bias[col];
            if (act)  v = tanhf(v);
            C[(size_t)(rbase + r) * Nd + col] = v;
        }
    }
}

// =====================================================================
// state init: hidden = repeat(dec_hidden, BEAM); scores; dec_input=SOS
// =====================================================================
__global__ void init_k(const float* __restrict__ dec_hidden,
                       float* __restrict__ hidden,
                       float* __restrict__ seq_scores,
                       int* __restrict__ dec_input)
{
    int idx = blockIdx.x * blockDim.x + threadIdx.x;   // NROWS*HDIM
    int n = idx >> 10, h = idx & 1023;
    hidden[idx] = dec_hidden[(size_t)(n >> 3) * HDIM + h];
    if (idx < NROWS) {
        seq_scores[idx] = ((idx & 7) == 0) ? 0.0f : NEGF;
        dec_input[idx]  = SOS_ID;
    }
}

// embedding gather: x[n,e] = embedding[dec_input[n], e]
__global__ void embed_k(const float* __restrict__ emb,
                        const int* __restrict__ dec_in,
                        float* __restrict__ x)
{
    int idx = blockIdx.x * blockDim.x + threadIdx.x;   // NROWS*EDIM
    int n = idx >> 10, e = idx & 1023;
    x[idx] = emb[(size_t)dec_in[n] * EDIM + e];
}

// GRU elementwise combine
__global__ void gru_combine_k(const float* __restrict__ gx,
                              const float* __restrict__ gh,
                              const float* __restrict__ hold,
                              float* __restrict__ hnew)
{
    int idx = blockIdx.x * blockDim.x + threadIdx.x;   // NROWS*HDIM
    int n = idx >> 10, j = idx & 1023;
    const float* gxr = gx + (size_t)n * 3072;
    const float* ghr = gh + (size_t)n * 3072;
    float r  = 1.0f / (1.0f + expf(-(gxr[j]        + ghr[j])));
    float z  = 1.0f / (1.0f + expf(-(gxr[1024 + j] + ghr[1024 + j])));
    float nn = tanhf(gxr[2048 + j] + r * ghr[2048 + j]);
    hnew[idx] = (1.0f - z) * nn + z * hold[idx];
}

// attention scores: one wave32 per (n,l) dot product of length H
__global__ void attn_scores_k(const float* __restrict__ h,
                              const float* __restrict__ qenc,
                              const int* __restrict__ qlen,
                              float* __restrict__ att)
{
    int gw   = (blockIdx.x * blockDim.x + threadIdx.x) >> 5;
    int lane = threadIdx.x & 31;
    int n = gw >> 6, l = gw & 63;                      // LSEQ = 64
    const float* hr = h    + (size_t)n * HDIM;
    const float* qr = qenc + ((size_t)(n >> 3) * LSEQ + l) * HDIM;
    float acc = 0.0f;
    for (int k = lane; k < HDIM; k += 32) acc += hr[k] * qr[k];
    for (int off = 16; off; off >>= 1) acc += __shfl_xor(acc, off, 32);
    if (lane == 0) {
        float v = acc * 0.03125f;                       // 1/sqrt(1024)
        if (l >= qlen[n >> 3]) v = NEGF;
        att[n * LSEQ + l] = v;
    }
}

// softmax over L + context vector; one block per row n
__global__ void softmax_ctx_k(const float* __restrict__ att,
                              const float* __restrict__ qenc,
                              float* __restrict__ ctx)
{
    __shared__ float a[LSEQ];
    __shared__ float inv_s;
    int n = blockIdx.x, tid = threadIdx.x;
    if (tid < LSEQ) a[tid] = att[n * LSEQ + tid];
    __syncthreads();
    if (tid == 0) {
        float m = a[0];
        for (int l = 1; l < LSEQ; ++l) m = fmaxf(m, a[l]);
        float s = 0.0f;
        for (int l = 0; l < LSEQ; ++l) { a[l] = expf(a[l] - m); s += a[l]; }
        inv_s = 1.0f / s;
    }
    __syncthreads();
    if (tid < LSEQ) a[tid] *= inv_s;
    __syncthreads();
    const float* q = qenc + (size_t)(n >> 3) * LSEQ * HDIM;
    for (int hh = tid; hh < HDIM; hh += blockDim.x) {
        float acc = 0.0f;
        for (int l = 0; l < LSEQ; ++l) acc += a[l] * q[(size_t)l * HDIM + hh];
        ctx[(size_t)n * HDIM + hh] = acc;
    }
}

// hc = concat(h_new, ctx)
__global__ void concat_k(const float* __restrict__ hn,
                         const float* __restrict__ ctx,
                         float* __restrict__ hc)
{
    int idx = blockIdx.x * blockDim.x + threadIdx.x;   // NROWS*2048
    int n = idx >> 11, j = idx & 2047;
    hc[idx] = (j < HDIM) ? hn[(size_t)n * HDIM + j]
                         : ctx[(size_t)n * HDIM + (j - HDIM)];
}

// per-row max and log-sum-exp of logits
__global__ void rowstats_k(const float* __restrict__ logits,
                           float* __restrict__ rmax, float* __restrict__ rlse)
{
    __shared__ float red[256];
    int n = blockIdx.x, tid = threadIdx.x;
    const float* row = logits + (size_t)n * VDIM;
    float m = -3.4e38f;
    for (int v = tid; v < VDIM; v += 256) m = fmaxf(m, row[v]);
    red[tid] = m; __syncthreads();
    for (int s = 128; s; s >>= 1) { if (tid < s) red[tid] = fmaxf(red[tid], red[tid + s]); __syncthreads(); }
    m = red[0]; __syncthreads();
    float sum = 0.0f;
    for (int v = tid; v < VDIM; v += 256) sum += expf(row[v] - m);
    red[tid] = sum; __syncthreads();
    for (int s = 128; s; s >>= 1) { if (tid < s) red[tid] += red[tid + s]; __syncthreads(); }
    if (tid == 0) { rmax[n] = m; rlse[n] = logf(red[0]); }
}

// top-8 over BEAM*V candidates per batch (8 iterative arg-max passes)
__global__ void topk_k(const float* __restrict__ logits,
                       const float* __restrict__ rmax, const float* __restrict__ rlse,
                       float* __restrict__ seq_scores, int* __restrict__ dec_input,
                       float* __restrict__ step_scores, int* __restrict__ step_preds,
                       int* __restrict__ step_syms, int t)
{
    __shared__ float rv[256]; __shared__ int ri[256];
    __shared__ int sel[BEAMK]; __shared__ float selv[BEAMK];
    int b = blockIdx.x, tid = threadIdx.x;
    float base[BEAMK];
#pragma unroll
    for (int k = 0; k < BEAMK; ++k) {
        int row = b * BEAMK + k;
        base[k] = seq_scores[row] - rmax[row] - rlse[row];
    }
    for (int p = 0; p < BEAMK; ++p) {
        float bv = -3.4e38f; int bi = 0x7fffffff;
        for (int k = 0; k < BEAMK; ++k) {
            const float* row = logits + (size_t)(b * BEAMK + k) * VDIM;
            float bk = base[k];
            for (int v = tid; v < VDIM; v += 256) {
                int idx = k * VDIM + v;
                bool skip = false;
                for (int q = 0; q < p; ++q) if (sel[q] == idx) skip = true;
                if (skip) continue;
                float val = bk + row[v];
                if (val > bv || (val == bv && idx < bi)) { bv = val; bi = idx; }
            }
        }
        rv[tid] = bv; ri[tid] = bi; __syncthreads();
        for (int s = 128; s; s >>= 1) {
            if (tid < s) {
                if (rv[tid + s] > rv[tid] ||
                    (rv[tid + s] == rv[tid] && ri[tid + s] < ri[tid])) {
                    rv[tid] = rv[tid + s]; ri[tid] = ri[tid + s];
                }
            }
            __syncthreads();
        }
        if (tid == 0) { sel[p] = ri[0]; selv[p] = rv[0]; }
        __syncthreads();
    }
    if (tid < BEAMK) {
        int j = tid, idx = sel[j];
        int sym  = idx % VDIM;
        int pred = idx / VDIM + b * BEAMK;
        int nrow = b * BEAMK + j;
        step_scores[t * NROWS + nrow] = selv[j];
        step_preds [t * NROWS + nrow] = pred;
        step_syms  [t * NROWS + nrow] = sym;
        seq_scores[nrow] = (sym == EOS_ID) ? NEGF : selv[j];
        dec_input[nrow]  = sym;
    }
}

// hidden[n] = h_new[pred[n]]
__global__ void reorder_k(const float* __restrict__ hnew,
                          const int* __restrict__ step_preds,
                          float* __restrict__ hidden, int t)
{
    int idx = blockIdx.x * blockDim.x + threadIdx.x;   // NROWS*HDIM
    int n = idx >> 10, h = idx & 1023;
    hidden[idx] = hnew[(size_t)step_preds[t * NROWS + n] * HDIM + h];
}

// final per-batch sort of last-step scores + predecessor backtrack
__global__ void backtrack_k(const float* __restrict__ step_scores,
                            const int* __restrict__ step_preds,
                            const int* __restrict__ step_syms,
                            float* __restrict__ out, int* __restrict__ t_pred)
{
    int tid = threadIdx.x;                              // 128 threads, 1 block
    if (tid < BSZ) {
        int b = tid;
        float sc[BEAMK]; int id[BEAMK];
        for (int k = 0; k < BEAMK; ++k) {
            sc[k] = step_scores[(TMAX - 1) * NROWS + b * BEAMK + k];
            id[k] = k;
        }
        for (int i = 0; i < BEAMK; ++i) {               // selection sort desc (swap-based, not stable)
            int best = i;
            for (int j = i + 1; j < BEAMK; ++j) if (sc[j] > sc[best]) best = j;
            float ts = sc[i]; sc[i] = sc[best]; sc[best] = ts;
            int   ti = id[i]; id[i] = id[best]; id[best] = ti;
            out[b * BEAMK + i]    = sc[i];
            t_pred[b * BEAMK + i] = id[i] + b * BEAMK;
        }
    }
    __syncthreads();
    int n = tid;
    int carry = t_pred[n];
    for (int tt = TMAX - 1; tt >= 0; --tt) {
        int sym = step_syms[tt * NROWS + carry];
        out[NROWS + n * TMAX + tt] = (float)sym;
        carry = step_preds[tt * NROWS + carry];
    }
}

// =====================================================================
extern "C" void kernel_launch(void* const* d_in, const int* in_sizes, int n_in,
                              void* d_out, int out_size, void* d_ws, size_t ws_size,
                              hipStream_t stream)
{
    const float* dec_hidden = (const float*)d_in[0];
    const float* q_enc      = (const float*)d_in[1];
    const int*   q_len      = (const int*)  d_in[2];
    const float* embedding  = (const float*)d_in[3];
    const float* W_ih       = (const float*)d_in[4];
    const float* W_hh       = (const float*)d_in[5];
    const float* b_ih       = (const float*)d_in[6];
    const float* b_hh       = (const float*)d_in[7];
    const float* W_c        = (const float*)d_in[8];
    const float* W_out      = (const float*)d_in[9];
    const float* b_out      = (const float*)d_in[10];
    float* out = (float*)d_out;

    char* ws = (char*)d_ws;
    size_t off = 0;
    auto carve = [&](size_t bytes) -> char* {
        char* p = ws + off;
        off += (bytes + 255) & ~(size_t)255;
        return p;
    };
    float* hidden      = (float*)carve((size_t)NROWS * HDIM * 4);
    float* h_new       = (float*)carve((size_t)NROWS * HDIM * 4);
    float* x_emb       = (float*)carve((size_t)NROWS * EDIM * 4);
    float* gx          = (float*)carve((size_t)NROWS * 3 * HDIM * 4);
    float* gh          = (float*)carve((size_t)NROWS * 3 * HDIM * 4);
    float* hc          = (float*)carve((size_t)NROWS * 2 * HDIM * 4);
    float* ctx         = (float*)carve((size_t)NROWS * HDIM * 4);
    float* att         = (float*)carve((size_t)NROWS * LSEQ * 4);
    float* logits      = (float*)carve((size_t)NROWS * VDIM * 4);
    float* rmax        = (float*)carve(NROWS * 4);
    float* rlse        = (float*)carve(NROWS * 4);
    float* seq_scores  = (float*)carve(NROWS * 4);
    int*   dec_input   = (int*)  carve(NROWS * 4);
    float* step_scores = (float*)carve((size_t)TMAX * NROWS * 4);
    int*   step_preds  = (int*)  carve((size_t)TMAX * NROWS * 4);
    int*   step_syms   = (int*)  carve((size_t)TMAX * NROWS * 4);
    int*   t_pred      = (int*)  carve(NROWS * 4);
    (void)ws_size; (void)in_sizes; (void)n_in; (void)out_size;

    init_k<<<(NROWS * HDIM) / 256, 256, 0, stream>>>(dec_hidden, hidden, seq_scores, dec_input);

    for (int t = 0; t < TMAX; ++t) {
        embed_k<<<(NROWS * EDIM) / 256, 256, 0, stream>>>(embedding, dec_input, x_emb);

        gemm_wmma_bf16<<<dim3(3 * HDIM / 256, NROWS / 16), 128, 0, stream>>>(
            x_emb, W_ih, b_ih, gx, EDIM, 3 * HDIM, 0);
        gemm_wmma_bf16<<<dim3(3 * HDIM / 256, NROWS / 16), 128, 0, stream>>>(
            hidden, W_hh, b_hh, gh, HDIM, 3 * HDIM, 0);

        gru_combine_k<<<(NROWS * HDIM) / 256, 256, 0, stream>>>(gx, gh, hidden, h_new);

        attn_scores_k<<<(NROWS * LSEQ * 32) / 256, 256, 0, stream>>>(h_new, q_enc, q_len, att);
        softmax_ctx_k<<<NROWS, 256, 0, stream>>>(att, q_enc, ctx);

        concat_k<<<(NROWS * 2 * HDIM) / 256, 256, 0, stream>>>(h_new, ctx, hc);
        gemm_wmma_bf16<<<dim3(HDIM / 256, NROWS / 16), 128, 0, stream>>>(
            hc, W_c, nullptr, x_emb /*reuse as o*/, 2 * HDIM, HDIM, 1);

        gemm_wmma_bf16<<<dim3(VDIM / 256, NROWS / 16), 128, 0, stream>>>(
            x_emb, W_out, b_out, logits, HDIM, VDIM, 0);

        rowstats_k<<<NROWS, 256, 0, stream>>>(logits, rmax, rlse);
        topk_k<<<BSZ, 256, 0, stream>>>(logits, rmax, rlse, seq_scores, dec_input,
                                        step_scores, step_preds, step_syms, t);
        reorder_k<<<(NROWS * HDIM) / 256, 256, 0, stream>>>(h_new, step_preds, hidden, t);
    }

    backtrack_k<<<1, 128, 0, stream>>>(step_scores, step_preds, step_syms, out, t_pred);
}
